// VectorQuantizer_85066122265065
// MI455X (gfx1250) — compile-verified, matrix operand order checked by reading, not run on hardware
//
#include <hip/hip_runtime.h>
#include <hip/hip_bf16.h>
#include <math.h>

// ---------------------------------------------------------------------------
// VQ-VAE quantizer forward for MI455X (gfx1250, wave32, WMMA).
// argmin_k ( ||e_k||^2 - 2 x.e_k )  -- ||x||^2 is a per-row constant and
// cannot change the argmin.  x.e^T runs on v_wmma_f32_16x16x32_bf16 (f32
// accumulate); the value-bearing outputs (quantized rows, loss) use the
// original f32 embedding, so only near-tie argmin picks can differ.
// ---------------------------------------------------------------------------

typedef __attribute__((ext_vector_type(16))) __bf16 v16bf;
typedef __attribute__((ext_vector_type(8)))  float  v8f;
typedef __attribute__((ext_vector_type(4)))  int    v4i;

#define N_PTS          131072
#define K_EMB          1024
#define D_DIM          64
#define ROWS_PER_BLOCK 128      // 8 waves x 16 rows
#define THREADS        256
#define N_CHUNKS       (K_EMB / 16)

// gfx1250 async global->LDS path (ASYNCcnt-tracked, bypasses VGPRs), with a
// compile-safe fallback if this toolchain doesn't declare the builtins.
#if defined(__gfx1250__) && \
    __has_builtin(__builtin_amdgcn_global_load_async_to_lds_b128) && \
    __has_builtin(__builtin_amdgcn_s_wait_asynccnt)
#define USE_ASYNC_LDS 1
typedef __attribute__((address_space(1))) v4i g_v4i;   // global (AS1)
typedef __attribute__((address_space(3))) v4i l_v4i;   // LDS (AS3)
#else
#define USE_ASYNC_LDS 0
#endif

// round-to-nearest-even f32 -> bf16 bit pattern
__device__ __forceinline__ unsigned short f32_to_bf16(float f) {
    unsigned u = __builtin_bit_cast(unsigned, f);
    unsigned r = u + 0x7FFFu + ((u >> 16) & 1u);
    return (unsigned short)(r >> 16);
}

// K-dim element index j (0..15) of a 16-bit 16x32 WMMA fragment for `lane`
// (ISA 7.12.2: lanes 0-15 hold K={0..7,16..23}, lanes 16-31 hold K={8..15,24..31})
__device__ __forceinline__ int dsel(int j, int lane) {
    return (j < 8 ? j : j + 8) + ((lane >= 16) ? 8 : 0);
}

// ---------------------------------------------------------------------------
// Kernel 1: pack embedding into bf16 B-fragment order, compute ||e_k||^2,
// zero the histogram counts.  65536 threads.
// wsB layout: frag fi = chunk*2 + kstep; element ((fi*32 + lane)*16 + j)
// ---------------------------------------------------------------------------
__global__ void vq_prep(const float* __restrict__ emb,
                        unsigned short* __restrict__ wsB,
                        float* __restrict__ wsN,
                        unsigned* __restrict__ counts) {
    int tid = blockIdx.x * blockDim.x + threadIdx.x;
    if (tid < K_EMB * D_DIM) {
        int j    = tid & 15;
        int lane = (tid >> 4) & 31;
        int s    = (tid >> 9) & 1;      // kstep: d 0..31 or 32..63
        int c    = tid >> 10;           // column chunk (16 embedding rows)
        int d    = s * 32 + dsel(j, lane);
        int row  = c * 16 + (lane & 15);
        wsB[tid] = f32_to_bf16(emb[row * D_DIM + d]);
    }
    if (tid < K_EMB) {
        const float* e = emb + tid * D_DIM;
        float s = 0.f;
#pragma unroll
        for (int d = 0; d < D_DIM; ++d) s += e[d] * e[d];
        wsN[tid]    = s;
        counts[tid] = 0u;
    }
}

// ---------------------------------------------------------------------------
// Kernel 2: main WMMA sweep + argmin + gather + partial loss.
// grid = N/128 blocks of 256 threads (8 waves).  Wave w owns rows
// blockRow0 + w*16 .. +15 and sweeps all K columns, two 16-col chunks per
// iteration (independent accumulators hide the WMMA->VALU XDL latency).
// ---------------------------------------------------------------------------
__global__ void __launch_bounds__(THREADS)
vq_main(const float* __restrict__ x, const float* __restrict__ emb,
        const unsigned short* __restrict__ wsB, const float* __restrict__ wsN,
        unsigned* __restrict__ counts, float* __restrict__ lossPartial,
        float* __restrict__ outQ, float* __restrict__ outIdx) {
    __shared__ float xs[ROWS_PER_BLOCK][D_DIM + 1];   // +1 pad kills bank conflicts
    __shared__ int   bestIdxLds[ROWS_PER_BLOCK];
    __shared__ float red[THREADS];

    const int tid       = threadIdx.x;
    const int lane      = tid & 31;
    const int wave      = tid >> 5;
    const int blockRow0 = blockIdx.x * ROWS_PER_BLOCK;

    // ---- stage x tile to LDS ----
#if USE_ASYNC_LDS
    // async DMA: 16B per lane per op, row-major with padded LDS rows.
    // 128 rows x 16 float4 = 2048 chunks / 256 threads = 8 per thread (uniform).
    for (int i = tid; i < ROWS_PER_BLOCK * (D_DIM / 4); i += THREADS) {
        int row = i >> 4;
        int c4  = i & 15;
        g_v4i* src = (g_v4i*)(x + (size_t)(blockRow0 + row) * D_DIM + c4 * 4);
        l_v4i* dst = (l_v4i*)&xs[row][c4 * 4];
        __builtin_amdgcn_global_load_async_to_lds_b128(src, dst, 0, 0);
    }
    __builtin_amdgcn_s_wait_asynccnt(0);
#else
    const float4* xg = reinterpret_cast<const float4*>(x + (size_t)blockRow0 * D_DIM);
    for (int i = tid; i < ROWS_PER_BLOCK * (D_DIM / 4); i += THREADS) {
        float4 v = xg[i];
        int row = i / (D_DIM / 4);
        int col = (i % (D_DIM / 4)) * 4;
        xs[row][col]     = v.x; xs[row][col + 1] = v.y;
        xs[row][col + 2] = v.z; xs[row][col + 3] = v.w;
    }
#endif
    __syncthreads();

    // ---- build this wave's two A fragments (16x32 bf16, d=0..31 / 32..63) ----
    const int rowBase = wave * 16;
    union AU { v16bf v; unsigned short s[16]; } a0, a1;
#pragma unroll
    for (int j = 0; j < 16; ++j) {
        int d = dsel(j, lane);
        int r = rowBase + (lane & 15);
        a0.s[j] = f32_to_bf16(xs[r][d]);
        a1.s[j] = f32_to_bf16(xs[r][d + 32]);
    }

    float best[8];
    int   bidx[8];
#pragma unroll
    for (int r = 0; r < 8; ++r) { best[r] = 3.4e38f; bidx[r] = K_EMB; }

    // ---- sweep all K, two 16-wide column chunks (4 WMMAs) per iteration ----
    const int ln15 = lane & 15;
    for (int c = 0; c < N_CHUNKS; c += 2) {
        const v16bf b00 = *reinterpret_cast<const v16bf*>(wsB + (((c * 2 + 0) * 32 + lane) << 4));
        const v16bf b01 = *reinterpret_cast<const v16bf*>(wsB + (((c * 2 + 1) * 32 + lane) << 4));
        const v16bf b10 = *reinterpret_cast<const v16bf*>(wsB + (((c * 2 + 2) * 32 + lane) << 4));
        const v16bf b11 = *reinterpret_cast<const v16bf*>(wsB + (((c * 2 + 3) * 32 + lane) << 4));
        if (c + 2 < N_CHUNKS)   // gfx1250 global_prefetch_b8 on the B frags after next
            __builtin_prefetch(wsB + (((c * 2 + 4) * 32 + lane) << 4), 0, 1);

        v8f acc0 = {};
        v8f acc1 = {};
        acc0 = __builtin_amdgcn_wmma_f32_16x16x32_bf16(false, a0.v, false, b00,
                                                       (short)0, acc0, false, false);
        acc0 = __builtin_amdgcn_wmma_f32_16x16x32_bf16(false, a1.v, false, b01,
                                                       (short)0, acc0, false, false);
        acc1 = __builtin_amdgcn_wmma_f32_16x16x32_bf16(false, a0.v, false, b10,
                                                       (short)0, acc1, false, false);
        acc1 = __builtin_amdgcn_wmma_f32_16x16x32_bf16(false, a1.v, false, b11,
                                                       (short)0, acc1, false, false);

        // consume chunk c while chunk c+1 drains the XDL pipe
        int   k0  = c * 16 + ln15;             // C col N = lane%16
        int   k1  = k0 + 16;
        float en0 = wsN[k0];
        float en1 = wsN[k1];
#pragma unroll
        for (int r = 0; r < 8; ++r) {          // C row M = r + 8*(lane/16)
            float s = fmaf(-2.0f, acc0[r], en0);
            if (s < best[r]) { best[r] = s; bidx[r] = k0; }
        }
#pragma unroll
        for (int r = 0; r < 8; ++r) {
            float s = fmaf(-2.0f, acc1[r], en1);
            if (s < best[r]) { best[r] = s; bidx[r] = k1; }
        }
    }

    // ---- argmin across the 16 lanes that share a row (tie -> lowest index) ----
#pragma unroll
    for (int r = 0; r < 8; ++r) {
#pragma unroll
        for (int m = 8; m >= 1; m >>= 1) {
            float os = __shfl_xor(best[r], m, 16);
            int   oi = __shfl_xor(bidx[r], m, 16);
            if (os < best[r] || (os == best[r] && oi < bidx[r])) {
                best[r] = os; bidx[r] = oi;
            }
        }
    }
    if ((lane & 15) == 0) {
        int half = lane >> 4;                  // lanes 0..15 -> rows r, 16..31 -> r+8
#pragma unroll
        for (int r = 0; r < 8; ++r)
            bestIdxLds[rowBase + half * 8 + r] = bidx[r];
    }
    __syncthreads();

    // ---- gather f32 embedding rows -> quantized output; accumulate MSE ----
    float lsum = 0.f;
    for (int e = tid; e < ROWS_PER_BLOCK * D_DIM; e += THREADS) {
        int row = e >> 6, col = e & 63;
        int k = bestIdxLds[row];
        float q = emb[k * D_DIM + col];
        outQ[(size_t)(blockRow0 + row) * D_DIM + col] = q;   // straight-through fwd == q
        float d = q - xs[row][col];
        lsum += d * d;
    }
    if (tid < ROWS_PER_BLOCK) {
        int k = bestIdxLds[tid];
        atomicAdd(&counts[k], 1u);
        outIdx[blockRow0 + tid] = (float)k;
    }

    red[tid] = lsum;
    __syncthreads();
    for (int s = THREADS / 2; s > 0; s >>= 1) {
        if (tid < s) red[tid] += red[tid + s];
        __syncthreads();
    }
    if (tid == 0) lossPartial[blockIdx.x] = red[0];
}

// ---------------------------------------------------------------------------
// Kernel 3: reduce loss partials + histogram entropy -> loss, perplexity.
// ---------------------------------------------------------------------------
__global__ void vq_finalize(const unsigned* __restrict__ counts,
                            const float* __restrict__ lossPartial,
                            float* __restrict__ lossOut,
                            float* __restrict__ perpOut) {
    __shared__ float sE[1024];
    __shared__ float sL[1024];
    int t = threadIdx.x;
    float p = (float)counts[t] * (1.0f / (float)N_PTS);
    sE[t] = p * logf(p + 1e-10f);
    sL[t] = lossPartial[t];
    __syncthreads();
    for (int s = 512; s > 0; s >>= 1) {
        if (t < s) { sE[t] += sE[t + s]; sL[t] += sL[t + s]; }
        __syncthreads();
    }
    if (t == 0) {
        *lossOut = 1.25f * sL[0] / (float)(N_PTS * D_DIM);   // diff2 + 0.25*diff2
        *perpOut = expf(-sE[0]);
    }
}

// ---------------------------------------------------------------------------
extern "C" void kernel_launch(void* const* d_in, const int* in_sizes, int n_in,
                              void* d_out, int out_size, void* d_ws, size_t ws_size,
                              hipStream_t stream) {
    const float* x   = (const float*)d_in[0];   // [N, D] f32
    const float* emb = (const float*)d_in[1];   // [K, D] f32
    float* out = (float*)d_out;

    // workspace: packed bf16 B frags | ||e||^2 | counts | loss partials
    char* ws = (char*)d_ws;
    unsigned short* wsB        = (unsigned short*)ws;            // 131072 B
    float*          wsN        = (float*)(ws + 131072);          //   4096 B
    unsigned*       counts     = (unsigned*)(ws + 135168);       //   4096 B
    float*          lossPartial= (float*)(ws + 139264);          //   4096 B

    // d_out layout (flat, return order): quantized_st [N*D], loss [1],
    // indices [N] (as float), perplexity [1]
    float* outQ    = out;
    float* lossOut = out + (size_t)N_PTS * D_DIM;
    float* outIdx  = lossOut + 1;
    float* perpOut = outIdx + N_PTS;

    vq_prep    <<<(K_EMB * D_DIM) / THREADS, THREADS, 0, stream>>>(emb, wsB, wsN, counts);
    vq_main    <<<N_PTS / ROWS_PER_BLOCK,    THREADS, 0, stream>>>(x, emb, wsB, wsN,
                                                                   counts, lossPartial,
                                                                   outQ, outIdx);
    vq_finalize<<<1, 1024, 0, stream>>>(counts, lossPartial, lossOut, perpOut);
}
